// BatteryMoEFlattenIntraCycleMoELayer_79310866088194
// MI455X (gfx1250) — matrix operand order repaired and checked by reading, not compiled, not run on hardware
//
#include <hip/hip_runtime.h>
#include <hip/hip_bf16.h>

typedef __attribute__((ext_vector_type(16))) __bf16 v16bf;
typedef __attribute__((ext_vector_type(8)))  __bf16 v8bf;
typedef __attribute__((ext_vector_type(8)))  float  v8f;

#define B_   128
#define L_   128
#define E_   8
#define INF_ 900
#define DM_  512
#define KT   32
#define KSTEPS 29      // ceil(900/32); last step ragged (4 valid k)
#define NTILE 128
#define MTILE 128

// ---------------- gating: masked softmax -> top-k -> renorm ----------------
// Also emits a compacted active-expert list so the GEMM kernel can run a
// flat, pipelined tile loop.
__global__ void moe_gating_kernel(const float* __restrict__ logits,
                                  const int*   __restrict__ masks,
                                  const int*   __restrict__ topk_ptr,
                                  float*       __restrict__ gates,  // [B,E]
                                  int*         __restrict__ nact,   // [B]
                                  int*         __restrict__ eid,    // [B,E]
                                  __bf16*      __restrict__ Y,
                                  long long out_size)
{
    int b = threadIdx.x;
    if (b < B_) {
        const int top_k = *topk_ptr;
        float lg[E_];
        float mx = -3.402823466e38f;
        #pragma unroll
        for (int e = 0; e < E_; ++e) { lg[e] = logits[b * E_ + e]; mx = fmaxf(mx, lg[e]); }
        float g[E_]; float s = 0.f;
        #pragma unroll
        for (int e = 0; e < E_; ++e) { g[e] = __expf(lg[e] - mx); s += g[e]; }
        #pragma unroll
        for (int e = 0; e < E_; ++e) { g[e] = (masks[b * E_ + e] == 1) ? (g[e] / s) : 0.f; }
        if (top_k > 0 && top_k < E_) {
            bool keep[E_];
            #pragma unroll
            for (int e = 0; e < E_; ++e) keep[e] = false;
            for (int t = 0; t < top_k; ++t) {
                int bi = -1; float bv = -1.f;
                #pragma unroll
                for (int e = 0; e < E_; ++e)
                    if (!keep[e] && g[e] > bv) { bv = g[e]; bi = e; }
                if (bi >= 0) keep[bi] = true;
            }
            #pragma unroll
            for (int e = 0; e < E_; ++e) if (!keep[e]) g[e] = 0.f;
        }
        float sg = 0.f;
        #pragma unroll
        for (int e = 0; e < E_; ++e) sg += g[e];
        const float inv = 1.f / (sg + 1e-9f);
        int cnt = 0;
        #pragma unroll
        for (int e = 0; e < E_; ++e) {
            const float gv = g[e] * inv;
            gates[b * E_ + e] = gv;
            if (gv != 0.f) { eid[b * E_ + cnt] = e; ++cnt; }
        }
        nact[b] = cnt;
    }
    // the two scalar (guide_loss, LB_loss) outputs are zero in eval mode
    if (threadIdx.x == 0) {
        long long base = (long long)B_ * L_ * DM_;
        for (long long i = base; i < out_size && i < base + 8; ++i) Y[i] = (__bf16)0.0f;
    }
}

// ---------------- grouped-GEMM main kernel (bf16 WMMA, f32 accum) ----------
__global__ __launch_bounds__(256)
void moe_gemm_kernel(const float* __restrict__ X,     // [B,L,INF]
                     const float* __restrict__ W,     // [E,INF,DM]
                     const float* __restrict__ bias,  // [E,DM]
                     const float* __restrict__ gates, // [B,E]
                     const int*   __restrict__ nact,  // [B]
                     const int*   __restrict__ eid,   // [B,E]
                     __bf16*      __restrict__ Y)     // [B,L,DM] bf16
{
    __shared__ __bf16 As[2][MTILE * KT];   // [m][k], K-contiguous
    __shared__ __bf16 Bs[2][NTILE * KT];   // [n][k], transposed, K-contiguous

    const int b    = blockIdx.y;
    const int n0   = blockIdx.x * NTILE;
    const int tid  = threadIdx.x;
    const int lane = tid & 31;
    const int wave = tid >> 5;
    const int wm   = (wave >> 2) * 64;  // wave M base (2 waves along M)
    const int wn   = (wave & 3) * 32;   // wave N base (4 waves along N)
    const int lm   = lane & 15;
    const int kp   = lane >> 4;

    // staging assignment (fixed per thread)
    const int arow = tid >> 1;            // A: row 0..127
    const int ac0  = (tid & 1) * 16;      // A: 16-col half
    const int bn   = (tid & 63) * 2;      // B: 2 columns
    const int bkq  = (tid >> 6) * 8;      // B: 8 k-values

    const float* Xb    = X + (size_t)b * L_ * INF_;
    const float* gw_b  = gates + b * E_;
    const int*   eid_b = eid + b * E_;
    const int    ne    = nact[b];         // block-uniform
    const int    total = ne * KSTEPS;

    v8f acc[4][2] = {};

    float ar[16];      // A staging regs (held across compute)
    float br[16];      // B staging regs: [0..7]=col bn, [8..15]=col bn+1

    auto load_tile = [&](int e_i, int s) {
        const int e  = eid_b[e_i];
        const float* We = W + (size_t)e * INF_ * DM_;
        const int k0 = s * KT;
        if (s < KSTEPS - 1) {
            // full tile: unguarded vector loads
            const float* asrc = Xb + (size_t)arow * INF_ + k0 + ac0;
            #pragma unroll
            for (int q = 0; q < 4; ++q) {
                const float4 v = *(const float4*)(asrc + q * 4);
                ar[q * 4 + 0] = v.x; ar[q * 4 + 1] = v.y;
                ar[q * 4 + 2] = v.z; ar[q * 4 + 3] = v.w;
            }
            #pragma unroll
            for (int i = 0; i < 8; ++i) {
                const float2 t2 = *(const float2*)(We + (size_t)(k0 + bkq + i) * DM_ + n0 + bn);
                br[i] = t2.x; br[8 + i] = t2.y;
            }
        } else {
            // ragged tail (k >= 900 -> 0), runs once per expert
            const float* asrc = Xb + (size_t)arow * INF_ + k0 + ac0;
            #pragma unroll
            for (int c = 0; c < 16; ++c)
                ar[c] = (k0 + ac0 + c < INF_) ? asrc[c] : 0.0f;
            #pragma unroll
            for (int i = 0; i < 8; ++i) {
                const int k = k0 + bkq + i;
                if (k < INF_) {
                    const float2 t2 = *(const float2*)(We + (size_t)k * DM_ + n0 + bn);
                    br[i] = t2.x; br[8 + i] = t2.y;
                } else {
                    br[i] = 0.f; br[8 + i] = 0.f;
                }
            }
        }
    };

    auto store_tile = [&](int buf, float ge) {
        v8bf p;
        #pragma unroll
        for (int q = 0; q < 8; ++q) p[q] = (__bf16)(ar[q] * ge);
        *(v8bf*)&As[buf][arow * KT + ac0] = p;
        #pragma unroll
        for (int q = 0; q < 8; ++q) p[q] = (__bf16)(ar[8 + q] * ge);
        *(v8bf*)&As[buf][arow * KT + ac0 + 8] = p;
        #pragma unroll
        for (int q = 0; q < 8; ++q) p[q] = (__bf16)br[q];
        *(v8bf*)&Bs[buf][bn * KT + bkq] = p;
        #pragma unroll
        for (int q = 0; q < 8; ++q) p[q] = (__bf16)br[8 + q];
        *(v8bf*)&Bs[buf][(bn + 1) * KT + bkq] = p;
    };

    auto compute = [&](int buf) {
        // A fragments: lane holds M=lm, K = kp*8+{0..7} then +16
        v16bf afrag[4];
        #pragma unroll
        for (int i = 0; i < 4; ++i) {
            const __bf16* arowp = &As[buf][(wm + i * 16 + lm) * KT];
            v8bf lo = *(const v8bf*)(arowp + kp * 8);
            v8bf hi = *(const v8bf*)(arowp + kp * 8 + 16);
            afrag[i] = __builtin_shufflevector(lo, hi,
                0,1,2,3,4,5,6,7,8,9,10,11,12,13,14,15);
        }
        // B fragments: lane holds N=lm, K = kp*8+{0..7} then +16
        v16bf bfrag[2];
        #pragma unroll
        for (int j = 0; j < 2; ++j) {
            const __bf16* browp = &Bs[buf][(wn + j * 16 + lm) * KT];
            v8bf lo = *(const v8bf*)(browp + kp * 8);
            v8bf hi = *(const v8bf*)(browp + kp * 8 + 16);
            bfrag[j] = __builtin_shufflevector(lo, hi,
                0,1,2,3,4,5,6,7,8,9,10,11,12,13,14,15);
        }
        #pragma unroll
        for (int i = 0; i < 4; ++i)
            #pragma unroll
            for (int j = 0; j < 2; ++j)
                acc[i][j] = __builtin_amdgcn_wmma_f32_16x16x32_bf16(
                    false, afrag[i], false, bfrag[j],
                    (short)0, acc[i][j], false, false);
    };

    if (total > 0) {
        // prime the pipeline with tile 0
        load_tile(0, 0);
        store_tile(0, gw_b[eid_b[0]]);
        int cur = 0;
        int ss = 0, ei = 0;   // indices of the tile currently in buf `cur`
        for (int it = 0; it < total; ++it) {
            __syncthreads();                         // buf[cur] staged, other buf free
            const bool have_next = (it + 1 < total);
            int s_n = ss + 1, e_n = ei;
            if (s_n == KSTEPS) { s_n = 0; ++e_n; }
            if (have_next) load_tile(e_n, s_n);      // global loads in flight...
            compute(cur);                            // ...while WMMAs run
            if (have_next) store_tile(1 - cur, gw_b[eid_b[e_n]]);
            ss = s_n; ei = e_n;
            cur ^= 1;
        }
    }

    // ---- epilogue: gated bias (depends only on column) + bf16 store
    float gb[2] = {0.f, 0.f};
    for (int t = 0; t < ne; ++t) {
        const int e = eid_b[t];
        const float ge = gw_b[e];
        #pragma unroll
        for (int j = 0; j < 2; ++j)
            gb[j] += ge * bias[e * DM_ + n0 + wn + j * 16 + lm];
    }
    __bf16* Yb = Y + (size_t)b * L_ * DM_;
    #pragma unroll
    for (int i = 0; i < 4; ++i) {
        #pragma unroll
        for (int j = 0; j < 2; ++j) {
            const int n = n0 + wn + j * 16 + lm;
            #pragma unroll
            for (int r = 0; r < 8; ++r) {
                const int m = wm + i * 16 + kp * 8 + r;  // C/D: vgpr r -> M=r+8*(lane>=16)
                Yb[(size_t)m * DM_ + n] = (__bf16)(acc[i][j][r] + gb[j]);
            }
        }
    }
}

extern "C" void kernel_launch(void* const* d_in, const int* in_sizes, int n_in,
                              void* d_out, int out_size, void* d_ws, size_t ws_size,
                              hipStream_t stream) {
    const float* X      = (const float*)d_in[0];   // cycle_curve_data [B,L,900]
    const float* logits = (const float*)d_in[1];   // [B,E]
    const int*   masks  = (const int*)d_in[2];     // [B,E]
    const float* W      = (const float*)d_in[3];   // [E,900,512]
    const float* bias   = (const float*)d_in[4];   // [E,512]
    const int*   topk   = (const int*)d_in[5];     // scalar

    // workspace layout: gates [B,E] f32 | nact [B] i32 | eid [B,E] i32
    float* gw   = (float*)d_ws;
    int*   nact = (int*)((char*)d_ws + B_ * E_ * sizeof(float));
    int*   eid  = (int*)((char*)d_ws + B_ * E_ * sizeof(float) + B_ * sizeof(int));
    __bf16* Y   = (__bf16*)d_out;

    moe_gating_kernel<<<1, 128, 0, stream>>>(logits, masks, topk, gw, nact, eid, Y,
                                             (long long)out_size);
    dim3 grid(DM_ / NTILE, B_);
    moe_gemm_kernel<<<grid, 256, 0, stream>>>(X, W, bias, gw, nact, eid, Y);
}
